// RelativeDecoder_26534307955242
// MI455X (gfx1250) — matrix operand order repaired and checked by reading, not compiled, not run on hardware
//
#include <hip/hip_runtime.h>
#include <hip/hip_bf16.h>

typedef __attribute__((ext_vector_type(16))) __bf16 v16bf;
typedef __attribute__((ext_vector_type(8)))  float  v8f;

#define BTOT   262144
#define HDIM   128
#define SDIM   128
#define PRED   12
#define WAVES  8
#define THREADS (WAVES * 32)
#define ROWS_PER_BLOCK (WAVES * 16)
#define NBLOCKS (BTOT / ROWS_PER_BLOCK)

// ---- LDS layout (ushort element offsets for bf16 weight regions) ----
#define OFF_WIH   0
#define OFF_WHH   (OFF_WIH + 384 * 128)
#define OFF_W1    (OFF_WHH + 384 * 128)
#define OFF_WEMB  (OFF_W1  + 64 * 256)
#define OFF_W2    (OFF_WEMB + 128 * 32)
#define USHORT_TOTAL (OFF_W2 + 16 * 64)
#define BIAS_BYTE_OFF (USHORT_TOTAL * 2)
// float offsets within bias region
#define BRZ   0
#define BNI   256
#define BNH   384
#define BB1   512
#define BB2   576
#define BEMB  580
#define BIAS_FLOATS 708
#define BUF_BYTE_OFF (BIAS_BYTE_OFF + BIAS_FLOATS * 4)
#define PERWAVE_USH (2048 + 2048 + 512)  // buf_h(16x128) + buf_x(16x128) + buf_obs(16x32)
#define ND_BYTE_OFF (BUF_BYTE_OFF + WAVES * PERWAVE_USH * 2)
#define SMEM_BYTES  (ND_BYTE_OFF + WAVES * 64 * 4)   // + per-wave buf_nd(16x4 f32)

// ---- native bf16 converts (single v_cvt on gfx1250) ----
__device__ __forceinline__ unsigned short f2bf(float f) {
  __bf16 b = (__bf16)f;
  return __builtin_bit_cast(unsigned short, b);
}
__device__ __forceinline__ float bf2f(unsigned short s) {
  return (float)__builtin_bit_cast(__bf16, s);
}

__device__ __forceinline__ float fast_sigmoid(float x) {
  return 1.0f / (1.0f + __expf(-x));
}
__device__ __forceinline__ float fast_tanh(float x) {
#if __has_builtin(__builtin_amdgcn_tanhf)
  return __builtin_amdgcn_tanhf(x);
#elif __has_builtin(__builtin_amdgcn_tanh_f32)
  return __builtin_amdgcn_tanh_f32(x);
#else
  float e = __expf(-2.0f * x);
  return (1.0f - e) / (1.0f + e);
#endif
}

union FragCast {
  unsigned short u[16];
  uint4 q[2];
  v16bf bf;
};

// 32 contiguous bytes (16 bf16) from LDS -> one B/A fragment lane-chunk
__device__ __forceinline__ v16bf ld_frag32(const unsigned short* p) {
  FragCast f;
  f.q[0] = *(const uint4*)(p);
  f.q[1] = *(const uint4*)(p + 8);
  return f.bf;
}

// A-matrix fragment (16xK tile, 16-bit layout): lane m = lane&15,
// chunk base c = 8*(lane>>4); VGPR0-3 <- K=c..c+7, VGPR4-7 <- K=c+16..c+23
__device__ __forceinline__ v16bf ld_afrag(const unsigned short* buf, int rowStride,
                                          int kbase, int lane) {
  int m = lane & 15;
  int c = (lane >> 4) << 3;
  const unsigned short* p = buf + m * rowStride + kbase + c;
  FragCast f;
  f.q[0] = *(const uint4*)(p);
  f.q[1] = *(const uint4*)(p + 16);
  return f.bf;
}

// B-matrix fragment (KxN=32x16): lane n = lane&15 column, c = 8*(lane>>4): K = kbase+c..+7, kbase+c+16..+23
// weights stored (N,K) row-major -> two 16B chunks per lane
__device__ __forceinline__ v16bf ld_bfrag(const unsigned short* w, int kStride,
                                          int ntile, int kbase, int lane) {
  int n = (ntile << 4) + (lane & 15);
  int k = kbase + ((lane >> 4) << 3);
  const unsigned short* p = w + n * kStride + k;
  FragCast f;
  f.q[0] = *(const uint4*)(p);
  f.q[1] = *(const uint4*)(p + 16);
  return f.bf;
}

__device__ __forceinline__ v8f splat8(float v) {
  v8f r;
#pragma unroll
  for (int i = 0; i < 8; ++i) r[i] = v;
  return r;
}

__device__ __forceinline__ v8f wmma_bf16(v16bf a, v16bf b, v8f c) {
  return __builtin_amdgcn_wmma_f32_16x16x32_bf16(false, a, false, b, (short)0, c,
                                                 false, false);
}

__device__ __forceinline__ void wait_ds() {
  asm volatile("s_wait_dscnt 0" ::: "memory");
}

__global__ __launch_bounds__(THREADS, 1) void relative_decoder_kernel(
    const float* __restrict__ xy, const float* __restrict__ dxdy,
    const float* __restrict__ social, const float* __restrict__ h0,
    const float* __restrict__ embW, const float* __restrict__ embB,
    const float* __restrict__ Wih, const float* __restrict__ Whh,
    const float* __restrict__ bih, const float* __restrict__ bhh,
    const float* __restrict__ W1, const float* __restrict__ b1,
    const float* __restrict__ W2, const float* __restrict__ b2,
    float* __restrict__ out) {
  extern __shared__ char smemRaw[];
  unsigned short* smemU = (unsigned short*)smemRaw;
  unsigned short* w_ih  = smemU + OFF_WIH;
  unsigned short* w_hh  = smemU + OFF_WHH;
  unsigned short* w1s   = smemU + OFF_W1;
  unsigned short* wemb  = smemU + OFF_WEMB;
  unsigned short* w2s   = smemU + OFF_W2;
  float* biasF = (float*)(smemRaw + BIAS_BYTE_OFF);
  unsigned short* bufBase = (unsigned short*)(smemRaw + BUF_BYTE_OFF);
  float* ndBase = (float*)(smemRaw + ND_BYTE_OFF);

  const int tid  = threadIdx.x;
  const int lane = tid & 31;
  const int wave = tid >> 5;
  const int nl = lane & 15;
  const int g  = lane >> 4;

  // ---- stage weights (f32 global -> bf16 LDS), cooperatively ----
  for (int i = tid; i < 384 * 128; i += THREADS) w_ih[i] = f2bf(Wih[i]);
  for (int i = tid; i < 384 * 128; i += THREADS) w_hh[i] = f2bf(Whh[i]);
  for (int i = tid; i < 64 * 256; i += THREADS)  w1s[i]  = f2bf(W1[i]);
  for (int i = tid; i < 128 * 32; i += THREADS) {
    int n = i >> 5, k = i & 31;
    wemb[i] = (k < 8) ? f2bf(embW[n * 8 + k]) : (unsigned short)0;
  }
  for (int i = tid; i < 16 * 64; i += THREADS) {
    int n = i >> 6, k = i & 63;
    w2s[i] = (n < 4) ? f2bf(W2[n * 64 + k]) : (unsigned short)0;
  }
  for (int i = tid; i < 256; i += THREADS) biasF[BRZ + i] = bih[i] + bhh[i];
  for (int i = tid; i < 128; i += THREADS) {
    biasF[BNI + i] = bih[256 + i];
    biasF[BNH + i] = bhh[256 + i];
    biasF[BEMB + i] = embB[i];
  }
  for (int i = tid; i < 64; i += THREADS) biasF[BB1 + i] = b1[i];
  for (int i = tid; i < 4; i += THREADS)  biasF[BB2 + i] = b2[i];
  __syncthreads();

  // ---- per-wave staging buffers ----
  unsigned short* buf_h   = bufBase + wave * PERWAVE_USH;
  unsigned short* buf_x   = buf_h + 2048;
  unsigned short* buf_obs = buf_x + 2048;
  float* buf_nd = ndBase + wave * 64;

  const int rowBase = blockIdx.x * ROWS_PER_BLOCK + wave * 16;

  // social feature A-fragments, resident in registers for all 12 steps
  v16bf socf[4];
  {
    const float* srow = social + (size_t)(rowBase + nl) * SDIM;
    int c = g << 3;
#pragma unroll
    for (int kt = 0; kt < 4; ++kt) {
      FragCast f;
#pragma unroll
      for (int j = 0; j < 8; ++j) f.u[j]     = f2bf(srow[kt * 32 + c + j]);
#pragma unroll
      for (int j = 0; j < 8; ++j) f.u[8 + j] = f2bf(srow[kt * 32 + c + 16 + j]);
      socf[kt] = f.bf;
    }
  }

  // stage h0 -> buf_h (bf16)
  for (int i = lane; i < 16 * HDIM; i += 32)
    buf_h[i] = f2bf(h0[(size_t)(rowBase + (i >> 7)) * HDIM + (i & 127)]);

  // zero obs buffer (cols 8..31 stay zero forever = K-padding for emb GEMM)
  for (int i = lane; i < 16 * 32; i += 32) buf_obs[i] = 0;

  float last[8];
  if (lane < 16) {
    float4 a = *(const float4*)(xy + (size_t)(rowBase + lane) * 4);
    float4 b = *(const float4*)(dxdy + (size_t)(rowBase + lane) * 4);
    last[0] = a.x; last[1] = a.y; last[2] = a.z; last[3] = a.w;
    last[4] = b.x; last[5] = b.y; last[6] = b.z; last[7] = b.w;
#pragma unroll
    for (int j = 0; j < 8; ++j) buf_obs[lane * 32 + j] = f2bf(last[j]);
  }

  float* out0 = out;
  float* out1 = out + (size_t)PRED * BTOT * 4;

#pragma unroll 1
  for (int t = 0; t < PRED; ++t) {
    wait_ds();
    // ---- spatial embedding: x = last_obsv(16x8,pad32) @ embW^T -> buf_x (16x128 bf16)
    v16bf aObs = ld_afrag(buf_obs, 32, 0, lane);
#pragma unroll
    for (int nt = 0; nt < 8; ++nt) {
      v8f acc = splat8(biasF[BEMB + nt * 16 + nl]);
      acc = wmma_bf16(aObs, ld_bfrag(wemb, 32, nt, 0, lane), acc);
#pragma unroll
      for (int r = 0; r < 8; ++r)
        buf_x[(r + (g << 3)) * 128 + nt * 16 + nl] = f2bf(acc[r]);
    }
    wait_ds();

    // ---- GRU: load A-fragments of x and h (old) ----
    v16bf xf[4], hf[4];
#pragma unroll
    for (int kt = 0; kt < 4; ++kt) {
      xf[kt] = ld_afrag(buf_x, 128, kt * 32, lane);
      hf[kt] = ld_afrag(buf_h, 128, kt * 32, lane);
    }
#pragma unroll
    for (int nt = 0; nt < 8; ++nt) {
      v8f aR  = splat8(biasF[BRZ + nt * 16 + nl]);
      v8f aZ  = splat8(biasF[BRZ + 128 + nt * 16 + nl]);
      v8f aNi = splat8(biasF[BNI + nt * 16 + nl]);
      v8f aNh = splat8(biasF[BNH + nt * 16 + nl]);
#pragma unroll
      for (int kt = 0; kt < 4; ++kt) {
        int kb = kt * 32;
        aR  = wmma_bf16(xf[kt], ld_bfrag(w_ih, 128, nt,      kb, lane), aR);
        aR  = wmma_bf16(hf[kt], ld_bfrag(w_hh, 128, nt,      kb, lane), aR);
        aZ  = wmma_bf16(xf[kt], ld_bfrag(w_ih, 128, nt + 8,  kb, lane), aZ);
        aZ  = wmma_bf16(hf[kt], ld_bfrag(w_hh, 128, nt + 8,  kb, lane), aZ);
        aNi = wmma_bf16(xf[kt], ld_bfrag(w_ih, 128, nt + 16, kb, lane), aNi);
        aNh = wmma_bf16(hf[kt], ld_bfrag(w_hh, 128, nt + 16, kb, lane), aNh);
      }
      // gate math + h update (in-place on buf_h; old h frags already in regs)
#pragma unroll
      for (int r = 0; r < 8; ++r) {
        int idx = (r + (g << 3)) * 128 + nt * 16 + nl;
        float hold = bf2f(buf_h[idx]);
        float rg = fast_sigmoid(aR[r]);
        float zg = fast_sigmoid(aZ[r]);
        float ng = fast_tanh(aNi[r] + rg * aNh[r]);
        buf_h[idx] = f2bf((1.0f - zg) * ng + zg * hold);
      }
    }
    wait_ds();

    // ---- MLP layer 1: z1 = leaky_relu([h, social] @ W1^T + b1) -> buf_x (16x64)
    v16bf hn[4];
#pragma unroll
    for (int kt = 0; kt < 4; ++kt) hn[kt] = ld_afrag(buf_h, 128, kt * 32, lane);
#pragma unroll
    for (int nt = 0; nt < 4; ++nt) {
      v8f acc = splat8(biasF[BB1 + nt * 16 + nl]);
#pragma unroll
      for (int kt = 0; kt < 4; ++kt)
        acc = wmma_bf16(hn[kt], ld_bfrag(w1s, 256, nt, kt * 32, lane), acc);
#pragma unroll
      for (int kt = 0; kt < 4; ++kt)
        acc = wmma_bf16(socf[kt], ld_bfrag(w1s, 256, nt, 128 + kt * 32, lane), acc);
#pragma unroll
      for (int r = 0; r < 8; ++r) {
        float v = acc[r];
        v = (v > 0.0f) ? v : 0.01f * v;
        buf_x[(r + (g << 3)) * 64 + nt * 16 + nl] = f2bf(v);
      }
    }
    wait_ds();

    // ---- MLP layer 2: nd = leaky_relu(z1 @ W2^T + b2), N padded 4->16
    {
      v16bf z0 = ld_afrag(buf_x, 64, 0, lane);
      v16bf z1 = ld_afrag(buf_x, 64, 32, lane);
      v8f acc = splat8((nl < 4) ? biasF[BB2 + nl] : 0.0f);
      acc = wmma_bf16(z0, ld_bfrag(w2s, 64, 0, 0, lane), acc);
      acc = wmma_bf16(z1, ld_bfrag(w2s, 64, 0, 32, lane), acc);
      if (nl < 4) {
#pragma unroll
        for (int r = 0; r < 8; ++r) {
          float v = acc[r];
          v = (v > 0.0f) ? v : 0.01f * v;
          buf_nd[(r + (g << 3)) * 4 + nl] = v;
        }
      }
    }
    wait_ds();

    // ---- elementwise trajectory update + output store (lanes 0..15 own 1 row each)
    if (lane < 16) {
      int m = lane;
      float nd0 = buf_nd[m * 4 + 0], nd1 = buf_nd[m * 4 + 1];
      float nd2 = buf_nd[m * 4 + 2], nd3 = buf_nd[m * 4 + 3];
      float whx = last[2] * __expf(nd2);
      float why = last[3] * __expf(nd3);
      float xn = last[0] + nd0 - 0.5f * whx;
      float yn = last[1] + nd1 - 0.5f * why;
      size_t row = (size_t)rowBase + m;
      size_t off = ((size_t)t * BTOT + row) * 4;
      *(float4*)(out0 + off) = make_float4(xn, yn, whx, why);
      *(float4*)(out1 + off) = make_float4(nd0, nd1, nd2, nd3);
      last[0] = xn;  last[1] = yn;  last[2] = whx; last[3] = why;
      last[4] = nd0; last[5] = nd1; last[6] = nd2; last[7] = nd3;
#pragma unroll
      for (int j = 0; j < 8; ++j) buf_obs[m * 32 + j] = f2bf(last[j]);
    }
  }
}

extern "C" void kernel_launch(void* const* d_in, const int* in_sizes, int n_in,
                              void* d_out, int out_size, void* d_ws, size_t ws_size,
                              hipStream_t stream) {
  const float* xy     = (const float*)d_in[0];
  const float* dxdy   = (const float*)d_in[1];
  const float* social = (const float*)d_in[2];
  const float* h0     = (const float*)d_in[3];
  // d_in[4] = pred_len scalar (constant 12, baked in)
  const float* embW = (const float*)d_in[5];
  const float* embB = (const float*)d_in[6];
  const float* Wih  = (const float*)d_in[7];
  const float* Whh  = (const float*)d_in[8];
  const float* bih  = (const float*)d_in[9];
  const float* bhh  = (const float*)d_in[10];
  const float* W1   = (const float*)d_in[11];
  const float* b1   = (const float*)d_in[12];
  const float* W2   = (const float*)d_in[13];
  const float* b2   = (const float*)d_in[14];
  float* out = (float*)d_out;

  (void)in_sizes; (void)n_in; (void)out_size; (void)d_ws; (void)ws_size;

  hipFuncSetAttribute((const void*)relative_decoder_kernel,
                      hipFuncAttributeMaxDynamicSharedMemorySize, SMEM_BYTES);
  relative_decoder_kernel<<<NBLOCKS, THREADS, SMEM_BYTES, stream>>>(
      xy, dxdy, social, h0, embW, embB, Wih, Whh, bih, bhh, W1, b1, W2, b2, out);
}
